// ConvModel_62715112456465
// MI455X (gfx1250) — compile-verified, matrix operand order checked by reading, not run on hardware
//
#include <hip/hip_runtime.h>

// CDNA5 / gfx1250 implicit-GEMM 3x3 conv, bf16x3 split precision:
//   fp32 v = hi(bf16) + lo(bf16);  A*B ~= Ah*Bh + Ah*Bl + Al*Bh  (fp32 accum)
// Matrix op: V_WMMA_F32_16X16X32_BF16 (8192 MACs/instr).
// Block (256 thr, 8 waves): 128 out-ch x 32 w x 4 out-rows from one staged
// 64ch x 6row x 34col window. Wave = 2 M-tiles (32 ch) x 32 w x 2 rows:
// each B fragment feeds 2 A-tiles (halves LDS reads/WMMA), each A fragment
// feeds 2 rows x 2 col-tiles (amortizes L2 weight traffic).

typedef __attribute__((ext_vector_type(8)))  float   v8f;
typedef __attribute__((ext_vector_type(2)))  __bf16  v2bf;
typedef __attribute__((ext_vector_type(8)))  __bf16  v8bf;
typedef __attribute__((ext_vector_type(16))) __bf16  v16bf;

#define CONV_N    16
#define CONV_CIN  64
#define CONV_H    224
#define CONV_W    224
#define CONV_COUT 128
#define SPT       32                   // spatial tile along W; 224 = 7*32
#define OUTR      4                    // output rows per block
#define ROWS      6                    // staged input rows (OUTR + 2 halo)
#define CSTR      72                   // LDS channel stride (64 padded; mult of 8)
#define PLANE     (ROWS * 34 * CSTR)   // one bf16 plane: 14688 elems
#define WK        576                  // K = 64 * 9
#define NWELEM    (CONV_COUT * WK)     // 73728 weights

__device__ __forceinline__ v16bf cat8(v8bf a, v8bf b) {
    return __builtin_shufflevector(a, b, 0,1,2,3,4,5,6,7,8,9,10,11,12,13,14,15);
}
__device__ __forceinline__ v16bf ldfrag(const __bf16* p) {
    return cat8(*(const v8bf*)p, *(const v8bf*)(p + 16));
}
__device__ __forceinline__ v8f wmma_bf16(v16bf a, v16bf b, v8f c) {
    return __builtin_amdgcn_wmma_f32_16x16x32_bf16(false, a, false, b,
                                                   (short)0, c, false, false);
}

// ---- prep: fp32 OIHW weights -> hi/lo bf16 planes in [o][rc][i] order ----
__global__ __launch_bounds__(256)
void prep_weights(const float* __restrict__ wgt, __bf16* __restrict__ whi,
                  __bf16* __restrict__ wlo) {
    const int idx = blockIdx.x * 256 + threadIdx.x;
    if (idx >= NWELEM) return;
    const int o  = idx / WK;
    const int k  = idx - o * WK;
    const int rc = k / CONV_CIN;       // kernel position (kh*3+kw)
    const int i  = k - rc * CONV_CIN;  // input channel
    const float v = wgt[((size_t)o * CONV_CIN + i) * 9 + rc];
    const __bf16 hi = (__bf16)v;
    whi[idx] = hi;
    wlo[idx] = (__bf16)(v - (float)hi);
}

__global__ __launch_bounds__(256)
void conv3x3_wmma_bf16x3(const float* __restrict__ x,
                         const __bf16* __restrict__ whi,  // [128][9][64]
                         const __bf16* __restrict__ wlo,
                         const float* __restrict__ bias,
                         float* __restrict__ out) {
    __shared__ __bf16 ldsb[2 * PLANE];   // hi plane + lo plane: 58,752 B

    const int tid   = threadIdx.x;
    const int bid   = blockIdx.x;
    const int wtile = bid % (CONV_W / SPT);
    const int tmp   = bid / (CONV_W / SPT);
    const int h0    = (tmp % (CONV_H / OUTR)) * OUTR;  // first of 4 output rows
    const int n     = tmp / (CONV_H / OUTR);
    const int w0    = wtile * SPT;

    // ---- stage 64ch x 6row x 34col window as hi/lo bf16, channel-minor ----
    for (int p = tid; p < 32 * ROWS * 34; p += 256) {   // channel pairs
        const int ci2 = p / (ROWS * 34);
        const int rem = p - ci2 * (ROWS * 34);
        const int r   = rem / 34;
        const int ww  = rem - r * 34;
        const int ci  = ci2 * 2;
        const int gh  = h0 + r - 1;
        const int gw  = w0 + ww - 1;
        float v0 = 0.0f, v1 = 0.0f;
        if ((unsigned)gh < (unsigned)CONV_H && (unsigned)gw < (unsigned)CONV_W) {
            const float* px = &x[(((size_t)n * CONV_CIN + ci) * CONV_H + gh) * CONV_W + gw];
            v0 = px[0];
            v1 = px[CONV_H * CONV_W];
        }
        const __bf16 h0b = (__bf16)v0, h1b = (__bf16)v1;
        const v2bf hi = {h0b, h1b};
        const v2bf lo = {(__bf16)(v0 - (float)h0b), (__bf16)(v1 - (float)h1b)};
        const int li = (r * 34 + ww) * CSTR + ci;
        *(v2bf*)&ldsb[li]         = hi;
        *(v2bf*)&ldsb[PLANE + li] = lo;
    }
    __syncthreads();

    // ---- wave mapping: mg = channel group (32 ch), rg = row group (2 rows) ----
    const int wave = tid >> 5;
    const int lane = tid & 31;
    const int m    = lane & 15;
    const int half = lane >> 4;
    const int mg   = wave & 3;         // channels [32*mg, 32*mg+32)
    const int rg   = wave >> 2;        // output rows h0 + 2*rg + {0,1}

    // A fragment bases for the two 16-ch M-tiles (hi and lo planes)
    const __bf16* pa_h0 = whi + (size_t)(32 * mg + m) * WK + 8 * half;
    const __bf16* pa_l0 = wlo + (size_t)(32 * mg + m) * WK + 8 * half;
    const __bf16* pa_h1 = pa_h0 + 16 * WK;
    const __bf16* pa_l1 = pa_l0 + 16 * WK;

    v8f acc[2][2][2];                  // [Mtile][row][Ntile]
    #pragma unroll
    for (int t = 0; t < 2; ++t)
        #pragma unroll
        for (int rw = 0; rw < 2; ++rw)
            #pragma unroll
            for (int nt = 0; nt < 2; ++nt) acc[t][rw][nt] = (v8f){};

    #pragma unroll 1
    for (int r = 0; r < 3; ++r) {
        // Prefetch next kernel-row weight slices toward WGP$ (global_prefetch_b8)
        if (r < 2) {
            __builtin_prefetch(pa_h0 + (r + 1) * 3 * CONV_CIN, 0, 3);
            __builtin_prefetch(pa_l0 + (r + 1) * 3 * CONV_CIN, 0, 3);
            __builtin_prefetch(pa_h1 + (r + 1) * 3 * CONV_CIN, 0, 3);
            __builtin_prefetch(pa_l1 + (r + 1) * 3 * CONV_CIN, 0, 3);
        }
        #pragma unroll 1
        for (int c = 0; c < 3; ++c) {
            #pragma unroll
            for (int chunk = 0; chunk < 2; ++chunk) {   // K-chunk = 32 channels
                const int ko = (r * 3 + c) * CONV_CIN + chunk * 32;
                const v16bf Ah0 = ldfrag(pa_h0 + ko);
                const v16bf Al0 = ldfrag(pa_l0 + ko);
                const v16bf Ah1 = ldfrag(pa_h1 + ko);
                const v16bf Al1 = ldfrag(pa_l1 + ko);

                #pragma unroll
                for (int rw = 0; rw < 2; ++rw) {  // out row h0+2*rg+rw taps staged row 2*rg+r+rw
                    const __bf16* lc =
                        &ldsb[((2 * rg + r + rw) * 34 + m + c) * CSTR + 8 * half] + chunk * 32;
                    const v16bf Bh0 = ldfrag(lc);
                    const v16bf Bl0 = ldfrag(lc + PLANE);
                    const v16bf Bh1 = ldfrag(lc + 16 * CSTR);
                    const v16bf Bl1 = ldfrag(lc + PLANE + 16 * CSTR);

                    acc[0][rw][0] = wmma_bf16(Ah0, Bh0, acc[0][rw][0]);
                    acc[0][rw][0] = wmma_bf16(Ah0, Bl0, acc[0][rw][0]);
                    acc[0][rw][0] = wmma_bf16(Al0, Bh0, acc[0][rw][0]);
                    acc[0][rw][1] = wmma_bf16(Ah0, Bh1, acc[0][rw][1]);
                    acc[0][rw][1] = wmma_bf16(Ah0, Bl1, acc[0][rw][1]);
                    acc[0][rw][1] = wmma_bf16(Al0, Bh1, acc[0][rw][1]);

                    acc[1][rw][0] = wmma_bf16(Ah1, Bh0, acc[1][rw][0]);
                    acc[1][rw][0] = wmma_bf16(Ah1, Bl0, acc[1][rw][0]);
                    acc[1][rw][0] = wmma_bf16(Al1, Bh0, acc[1][rw][0]);
                    acc[1][rw][1] = wmma_bf16(Ah1, Bh1, acc[1][rw][1]);
                    acc[1][rw][1] = wmma_bf16(Ah1, Bl1, acc[1][rw][1]);
                    acc[1][rw][1] = wmma_bf16(Al1, Bh1, acc[1][rw][1]);
                }
            }
        }
    }

    // ---- epilogue: bias + store. D layout: reg rr -> row rr + 8*half, col = m ----
    #pragma unroll
    for (int t = 0; t < 2; ++t) {
        #pragma unroll
        for (int rr = 0; rr < 8; ++rr) {
            const int o    = 32 * mg + 16 * t + half * 8 + rr;
            const float bv = bias[o];
            const int hrow = h0 + 2 * rg;
            const size_t obase =
                (((size_t)n * CONV_COUT + o) * CONV_H + hrow) * CONV_W + w0 + m;
            out[obase]               = acc[t][0][0][rr] + bv;
            out[obase + 16]          = acc[t][0][1][rr] + bv;
            out[obase + CONV_W]      = acc[t][1][0][rr] + bv;
            out[obase + CONV_W + 16] = acc[t][1][1][rr] + bv;
        }
    }
}

extern "C" void kernel_launch(void* const* d_in, const int* in_sizes, int n_in,
                              void* d_out, int out_size, void* d_ws, size_t ws_size,
                              hipStream_t stream) {
    (void)in_sizes; (void)n_in; (void)out_size; (void)ws_size;
    const float* x    = (const float*)d_in[0];
    const float* wgt  = (const float*)d_in[1];
    const float* bias = (const float*)d_in[2];
    float* out        = (float*)d_out;

    __bf16* whi = (__bf16*)d_ws;               // 73728 bf16
    __bf16* wlo = whi + NWELEM;                // 73728 bf16 (288 KB total in d_ws)

    prep_weights<<<(NWELEM + 255) / 256, 256, 0, stream>>>(wgt, whi, wlo);

    const int grid = CONV_N * (CONV_H / OUTR) * (CONV_W / SPT);   // 16*56*7 = 6272
    conv3x3_wmma_bf16x3<<<grid, 256, 0, stream>>>(x, whi, wlo, bias, out);
}
